// ResidualBiLSTMBlock_50062138802419
// MI455X (gfx1250) — matrix operand order repaired and checked by reading, not compiled, not run on hardware
//
#include <hip/hip_runtime.h>
#include <hip/hip_bf16.h>

// ---------------------------------------------------------------------------
// Residual BiLSTM block for MI455X (gfx1250, wave32, WMMA bf16).
//   B=64, T=1024, D=256, U=256, 4U=1024, out=[B,T,512] f32
// ---------------------------------------------------------------------------

#define BB   64
#define TT   1024
#define DD   256
#define UU   256
#define G4U  1024          // 4*U
#define OUTC 512           // 2*U
#define ROWS (BB * TT)     // 65536 flattened (b,t) rows

// --- gfx1250 feature probes (host pass sees these as absent -> no-ops) -----
#if __has_builtin(__builtin_amdgcn_global_load_async_to_lds_b128)
#define HAVE_ASYNC_LDS 1
#else
#define HAVE_ASYNC_LDS 0
#endif

// mask 0x7: ALU/VALU/SALU may cross; VMEM/DS/WMMA ordering is preserved,
// so prefetch loads stay ahead of the WMMAs without bloating live ranges.
#if __has_builtin(__builtin_amdgcn_sched_barrier)
#define SCHED_BARRIER() __builtin_amdgcn_sched_barrier(0x7)
#else
#define SCHED_BARRIER() do {} while (0)
#endif

typedef __attribute__((ext_vector_type(16))) __bf16 bf16x16;
typedef __attribute__((ext_vector_type(8)))  __bf16 bf16x8;
typedef __attribute__((ext_vector_type(8)))  float  f32x8;
typedef __attribute__((ext_vector_type(4)))  float  f32x4;
typedef __attribute__((ext_vector_type(4)))  int    i32x4;

// RNE float -> bf16 without relying on implicit conversion support
__device__ __forceinline__ __bf16 f2bf(float f) {
    union { float f; unsigned u; } v; v.f = f;
    unsigned r = v.u + 0x7FFFu + ((v.u >> 16) & 1u);
    unsigned short h = (unsigned short)(r >> 16);
    __bf16 o; __builtin_memcpy(&o, &h, 2);
    return o;
}

__device__ __forceinline__ float sigm(float x) { return 1.0f / (1.0f + __expf(-x)); }

// 16B global -> LDS copy: async DMA (ASYNCcnt) when available, else sync copy.
// Builtin prototype (from compiler diagnostic): (int4 AS1*, int4 AS3*, Imm, Imm)
__device__ __forceinline__ void async_cp16(const float *__restrict__ g, float *l) {
#if HAVE_ASYNC_LDS
    __builtin_amdgcn_global_load_async_to_lds_b128(
        (i32x4 __attribute__((address_space(1))) *)(size_t)g,
        (i32x4 __attribute__((address_space(3))) *)(i32x4 *)l, 0, 0);
#else
    *(f32x4 *)l = *(const f32x4 *)g;
#endif
}

template <int N>
__device__ __forceinline__ void wait_async() {
#if HAVE_ASYNC_LDS
#if __has_builtin(__builtin_amdgcn_s_wait_asynccnt)
    __builtin_amdgcn_s_wait_asynccnt(N);
#else
    asm volatile("s_wait_asynccnt %0" ::"i"(N) : "memory");
#endif
#endif
}

// ---------------------------------------------------------------------------
// A-fragment (16x32 bf16, row-major source, leading dim ld elements).
// lane<16 : row=lane,    elems 0..15 = K[kc*32+0..16)
// lane>=16: row=lane-16, elems 0..15 = K[kc*32+16..32)
// ---------------------------------------------------------------------------
template <typename PtrT>
__device__ __forceinline__ bf16x16 load_frag_a(PtrT base, int ld, int kc, int lane) {
    const int m    = lane & 15;
    const int half = lane >> 4;
    PtrT p = base + (size_t)m * ld + kc * 32;
    bf16x8 lo = *(const bf16x8 *)(p + 16 * half);
    bf16x8 hi = *(const bf16x8 *)(p + 16 * half + 8);
    bf16x16 a;
#pragma unroll
    for (int i = 0; i < 8; ++i) { a[i] = lo[i]; a[8 + i] = hi[i]; }
    return a;
}

// ---------------------------------------------------------------------------
// B-fragment (32x16 bf16) from Bt stored [N][K] row-major (i.e. B transposed).
// lane<16 : col=lane,    K[kc*32 .. +16) ;  lane>=16: col=lane-16, K[+16..+32)
// ---------------------------------------------------------------------------
__device__ __forceinline__ bf16x16 load_frag_b(const __bf16 *Bt, int colBase, int kc, int lane) {
    const int n    = lane & 15;
    const int half = lane >> 4;
    const __bf16 *p = Bt + (size_t)(colBase + n) * DD + kc * 32 + 16 * half;
    return *(const bf16x16 *)p;
}

// ---------------------------------------------------------------------------
// Conversion / transpose prep kernels
// ---------------------------------------------------------------------------
__global__ void __launch_bounds__(256)
k_f32_to_bf16(const float *__restrict__ src, __bf16 *__restrict__ dst, int n) {
    int i = blockIdx.x * blockDim.x + threadIdx.x;
    for (; i < n; i += gridDim.x * blockDim.x) dst[i] = f2bf(src[i]);
}

// src: [K][N] row-major f32  ->  dst: [N][K] row-major bf16
__global__ void __launch_bounds__(256)
k_transpose_cvt(const float *__restrict__ src, __bf16 *__restrict__ dst, int K, int N) {
    int i = blockIdx.x * blockDim.x + threadIdx.x;
    int total = K * N;
    for (; i < total; i += gridDim.x * blockDim.x) {
        int k = i / N, n = i - k * N;
        dst[(size_t)n * K + k] = f2bf(src[i]);
    }
}

// ---------------------------------------------------------------------------
// GEMM: C[M][N] = A(bf16 [M][256]) * B(bf16, given as Bt [N][256]) + bias
// One wave computes a 32x64 block (2 M-tiles x 4 N-tiles): 8 WMMAs per
// 12 b128 loads per k-step, software-pipelined one k-step ahead.
// blockDim = 512 (16 waves -> 4/SIMD); __launch_bounds__(512) raises the
// per-wave VGPR cap to ~256 so the double-buffered fragments never spill.
// ---------------------------------------------------------------------------
__global__ void __launch_bounds__(512)
k_gemm_bf16(const __bf16 *__restrict__ A, const __bf16 *__restrict__ Bt,
            const float *__restrict__ bias, float *__restrict__ C,
            int M, int N) {
    const int lane = threadIdx.x & 31;
    const int half = lane >> 4;
    const int nn   = lane & 15;
    const int gw   = blockIdx.x * (blockDim.x >> 5) + (threadIdx.x >> 5);
    const int nGroups = N >> 6;              // 64 cols per wave
    const int mblk = gw / nGroups;           // 32-row block index
    const int nblk = gw - mblk * nGroups;
    if (mblk >= (M >> 5)) return;

    f32x8 acc[2][4];
#pragma unroll
    for (int t = 0; t < 4; ++t) {
        float b0 = bias ? bias[nblk * 64 + t * 16 + nn] : 0.0f;
#pragma unroll
        for (int r = 0; r < 8; ++r) { acc[0][t][r] = b0; acc[1][t][r] = b0; }
    }

    const __bf16 *A0 = A + (size_t)(mblk * 32) * DD;
    const __bf16 *A1 = A0 + (size_t)16 * DD;

    // prologue: fragments for kc = 0
    bf16x16 a0 = load_frag_a(A0, DD, 0, lane);
    bf16x16 a1 = load_frag_a(A1, DD, 0, lane);
    bf16x16 bC[4];
#pragma unroll
    for (int t = 0; t < 4; ++t) bC[t] = load_frag_b(Bt, nblk * 64 + t * 16, 0, lane);

#pragma unroll
    for (int kc = 0; kc < 8; ++kc) {
        bf16x16 a0n, a1n, bN[4];
        if (kc < 7) {                       // prefetch next k-step
            a0n = load_frag_a(A0, DD, kc + 1, lane);
            a1n = load_frag_a(A1, DD, kc + 1, lane);
#pragma unroll
            for (int t = 0; t < 4; ++t) bN[t] = load_frag_b(Bt, nblk * 64 + t * 16, kc + 1, lane);
        }
        SCHED_BARRIER();                    // keep prefetches ahead of the WMMAs
#pragma unroll
        for (int t = 0; t < 4; ++t) {
            acc[0][t] = __builtin_amdgcn_wmma_f32_16x16x32_bf16(
                false, a0, false, bC[t], (short)0, acc[0][t], false, false);
            acc[1][t] = __builtin_amdgcn_wmma_f32_16x16x32_bf16(
                false, a1, false, bC[t], (short)0, acc[1][t], false, false);
        }
        if (kc < 7) {
            a0 = a0n; a1 = a1n;
#pragma unroll
            for (int t = 0; t < 4; ++t) bC[t] = bN[t];
        }
    }

    // streaming output (read once much later) -> nontemporal, keep L2 for weights
#pragma unroll
    for (int mt = 0; mt < 2; ++mt) {
#pragma unroll
        for (int t = 0; t < 4; ++t) {
            const int col = nblk * 64 + t * 16 + nn;
#pragma unroll
            for (int r = 0; r < 8; ++r) {
                const int row = mblk * 32 + mt * 16 + r + 8 * half;
                __builtin_nontemporal_store(acc[mt][t][r], &C[(size_t)row * N + col]);
            }
        }
    }
}

// ---------------------------------------------------------------------------
// LSTM scan.  grid = 8 blocks: dir = blockIdx&1, batch-slice = blockIdx>>1.
// blockDim = 512 (16 waves).  Wave w owns unit columns [16w,16w+16) across
// all 4 gates; cell state c stays in registers; h round-trips through a
// double-buffered LDS tile (one barrier per step).  x_proj tiles are DMA'd
// into a triple-buffered LDS region two steps ahead with async-to-LDS ops
// (ASYNCcnt), so the cold 512MB HBM stream never touches the serial chain.
// ---------------------------------------------------------------------------
#define HLD 264            // h row pitch (bf16): 256 + 8 pad
#define ZLD 1028           // z row pitch (f32): 1024 + 4 pad
#define H16 (16 * HLD)
#define Z16 (16 * ZLD)
#define SCAN_SMEM (2 * H16 * 2 + 3 * Z16 * 4)   // 16896 + 197376 = 214272 B

__global__ void __launch_bounds__(512)
k_lstm_scan(const __bf16 *__restrict__ Ut_f, const __bf16 *__restrict__ Ut_b,
            const float *__restrict__ zf, const float *__restrict__ zb,
            float *__restrict__ hf, float *__restrict__ hb) {
    extern __shared__ char smem[];
    __bf16 *hbuf = (__bf16 *)smem;                  // [2][16][HLD]
    float  *zls  = (float *)(smem + 2 * H16 * 2);   // [3][16][ZLD]

    const int dir = blockIdx.x & 1;
    const int b0  = (blockIdx.x >> 1) * 16;
    const __bf16 *Ut = dir ? Ut_b : Ut_f;
    const float  *Z  = dir ? zb   : zf;
    float        *H  = dir ? hb   : hf;

    const int lane = threadIdx.x & 31;
    const int half = lane >> 4;
    const int nn   = lane & 15;
    const int w    = threadIdx.x >> 5;   // 0..15: wave id == batch row it DMAs
    const int j0   = w * 16;

    // zero the h buffer read at step 0
    for (int i = threadIdx.x; i < H16; i += blockDim.x) hbuf[i] = f2bf(0.0f);

    // prologue: DMA z tiles for steps 0 and 1 (wave w copies batch row w)
    {
        const float *s0 = Z + ((size_t)(b0 + w) * TT + (dir ? TT - 1 : 0)) * G4U;
        const float *s1 = Z + ((size_t)(b0 + w) * TT + (dir ? TT - 2 : 1)) * G4U;
        float *d0 = zls + 0 * Z16 + w * ZLD;
        float *d1 = zls + 1 * Z16 + w * ZLD;
#pragma unroll
        for (int c = 0; c < 8; ++c) async_cp16(s0 + (lane + c * 32) * 4, d0 + (lane + c * 32) * 4);
#pragma unroll
        for (int c = 0; c < 8; ++c) async_cp16(s1 + (lane + c * 32) * 4, d1 + (lane + c * 32) * 4);
    }
    wait_async<8>();       // in-order per wave: buffer 0 complete
    __syncthreads();

    float cst[8];
#pragma unroll
    for (int r = 0; r < 8; ++r) cst[r] = 0.0f;

    for (int s = 0; s < TT; ++s) {
        const int t = dir ? (TT - 1 - s) : s;

        // C-init from LDS (low-latency ds reads; tile DMA'd two steps ago)
        const float *zr = zls + (s % 3) * Z16;
        f32x8 acc[4];
#pragma unroll
        for (int g = 0; g < 4; ++g)
#pragma unroll
            for (int r = 0; r < 8; ++r)
                acc[g][r] = zr[(r + 8 * half) * ZLD + g * 256 + j0 + nn];

        // kick off DMA for step s+2
        if (s + 2 < TT) {
            const int tn = dir ? (TT - 3 - s) : (s + 2);
            const float *sn = Z + ((size_t)(b0 + w) * TT + tn) * G4U;
            float *dn = zls + ((s + 2) % 3) * Z16 + w * ZLD;
#pragma unroll
            for (int c = 0; c < 8; ++c)
                async_cp16(sn + (lane + c * 32) * 4, dn + (lane + c * 32) * 4);
        }

        const __bf16 *hr = hbuf + (s & 1) * H16;          // read h_{t-1}
        __bf16       *hw = hbuf + ((s & 1) ^ 1) * H16;    // write h_t

        // opaque copy of Ut: stop LICM from hoisting 32 t-invariant B
        // fragments (256 VGPRs) out of the loop, which would break the
        // 16-wave/WGP residency this kernel requires.
        const __bf16 *Uts = Ut;
        asm volatile("" : "+v"(Uts));

        // z += h_{t-1} @ U_rec  (pipelined one k-step ahead)
        bf16x16 aC = load_frag_a(hr, HLD, 0, lane);
        bf16x16 bC[4];
#pragma unroll
        for (int g = 0; g < 4; ++g) bC[g] = load_frag_b(Uts, g * 256 + j0, 0, lane);

#pragma unroll
        for (int kc = 0; kc < 8; ++kc) {
            bf16x16 aN, bN[4];
            if (kc < 7) {
                aN = load_frag_a(hr, HLD, kc + 1, lane);
#pragma unroll
                for (int g = 0; g < 4; ++g) bN[g] = load_frag_b(Uts, g * 256 + j0, kc + 1, lane);
            }
            SCHED_BARRIER();
#pragma unroll
            for (int g = 0; g < 4; ++g)
                acc[g] = __builtin_amdgcn_wmma_f32_16x16x32_bf16(
                    false, aC, false, bC[g], (short)0, acc[g], false, false);
            if (kc < 7) {
                aC = aN;
#pragma unroll
                for (int g = 0; g < 4; ++g) bC[g] = bN[g];
            }
        }

        // gates (Keras order i, f, g, o) -> c, h ; write h to LDS + global
#pragma unroll
        for (int r = 0; r < 8; ++r) {
            const float ig = sigm(acc[0][r]);
            const float fg = sigm(acc[1][r]);
            const float gg = tanhf(acc[2][r]);
            const float og = sigm(acc[3][r]);
            cst[r] = fg * cst[r] + ig * gg;
            const float h = og * tanhf(cst[r]);
            const int m = r + 8 * half;
            hw[m * HLD + j0 + nn] = f2bf(h);
            __builtin_nontemporal_store(h, &H[((size_t)(b0 + m) * TT + t) * UU + j0 + nn]);
        }

        // buffer for s+1 must be resident before next step reads it:
        // the newest 8 async ops (for s+2) may stay in flight.
        if (s + 2 < TT) wait_async<8>(); else wait_async<0>();
        __syncthreads();
    }
}

// ---------------------------------------------------------------------------
// LayerNorm(concat(hf,hb)) * gamma + beta + residual  -> out
// one (b,t) row (512 values) per 256-thread block
// ---------------------------------------------------------------------------
__global__ void __launch_bounds__(256)
k_ln_residual(const float *__restrict__ hf, const float *__restrict__ hb,
              const float *__restrict__ resid,
              const float *__restrict__ gamma, const float *__restrict__ beta,
              float *__restrict__ out) {
    const int row = blockIdx.x;       // b*T + t
    const int tid = threadIdx.x;      // 0..255
    const float v0 = __builtin_nontemporal_load(&hf[(size_t)row * UU + tid]);
    const float v1 = __builtin_nontemporal_load(&hb[(size_t)row * UU + tid]);

    __shared__ float ssum[256], ssq[256];
    ssum[tid] = v0 + v1;
    ssq[tid]  = v0 * v0 + v1 * v1;
    __syncthreads();
#pragma unroll
    for (int s = 128; s > 0; s >>= 1) {
        if (tid < s) { ssum[tid] += ssum[tid + s]; ssq[tid] += ssq[tid + s]; }
        __syncthreads();
    }
    const float mu  = ssum[0] * (1.0f / OUTC);
    const float var = ssq[0] * (1.0f / OUTC) - mu * mu;
    const float rs  = rsqrtf(var + 1e-3f);

    const size_t ob = (size_t)row * OUTC;
    const float r0 = __builtin_nontemporal_load(&resid[ob + tid]);
    const float r1 = __builtin_nontemporal_load(&resid[ob + UU + tid]);
    out[ob + tid]      = (v0 - mu) * rs * gamma[tid]      + beta[tid]      + r0;
    out[ob + UU + tid] = (v1 - mu) * rs * gamma[UU + tid] + beta[UU + tid] + r1;
}

// ---------------------------------------------------------------------------
// Host launcher.  Workspace layout (bytes, all 256B aligned, ~802 MB total):
//   xb bf16 32MB | Wt_f/Wt_b/Ut_f/Ut_b bf16 4x0.5MB | Wpt bf16 0.25MB
//   zf/zb f32 2x256MB | hf/hb f32 2x64MB | resid f32 128MB
// ---------------------------------------------------------------------------
extern "C" void kernel_launch(void* const* d_in, const int* in_sizes, int n_in,
                              void* d_out, int out_size, void* d_ws, size_t ws_size,
                              hipStream_t stream) {
    (void)in_sizes; (void)n_in; (void)out_size; (void)ws_size;

    const float* x      = (const float*)d_in[0];
    const float* W_f    = (const float*)d_in[1];
    const float* U_f    = (const float*)d_in[2];
    const float* b_f    = (const float*)d_in[3];
    const float* W_b    = (const float*)d_in[4];
    const float* U_b    = (const float*)d_in[5];
    const float* b_b    = (const float*)d_in[6];
    const float* gamma  = (const float*)d_in[7];
    const float* beta   = (const float*)d_in[8];
    const float* W_proj = (const float*)d_in[9];
    float* out = (float*)d_out;

    char* ws = (char*)d_ws;
    size_t off = 0;
    __bf16* xb   = (__bf16*)(ws + off); off += (size_t)ROWS * DD * 2;
    __bf16* Wt_f = (__bf16*)(ws + off); off += (size_t)G4U * DD * 2;
    __bf16* Wt_b = (__bf16*)(ws + off); off += (size_t)G4U * DD * 2;
    __bf16* Ut_f = (__bf16*)(ws + off); off += (size_t)G4U * DD * 2;
    __bf16* Ut_b = (__bf16*)(ws + off); off += (size_t)G4U * DD * 2;
    __bf16* Wpt  = (__bf16*)(ws + off); off += (size_t)OUTC * DD * 2;
    float* zf    = (float*)(ws + off);  off += (size_t)ROWS * G4U * 4;
    float* zb    = (float*)(ws + off);  off += (size_t)ROWS * G4U * 4;
    float* hf    = (float*)(ws + off);  off += (size_t)ROWS * UU * 4;
    float* hb    = (float*)(ws + off);  off += (size_t)ROWS * UU * 4;
    float* resid = (float*)(ws + off);  off += (size_t)ROWS * OUTC * 4;

    // ---- prep: convert x, transpose+convert weights --------------------
    {
        const int n = ROWS * DD;
        k_f32_to_bf16<<<(n + 255) / 256, 256, 0, stream>>>(x, xb, n);
        const int wn = DD * G4U;
        k_transpose_cvt<<<(wn + 255) / 256, 256, 0, stream>>>(W_f,    Wt_f, DD, G4U);
        k_transpose_cvt<<<(wn + 255) / 256, 256, 0, stream>>>(W_b,    Wt_b, DD, G4U);
        k_transpose_cvt<<<(wn + 255) / 256, 256, 0, stream>>>(U_f,    Ut_f, UU, G4U);
        k_transpose_cvt<<<(wn + 255) / 256, 256, 0, stream>>>(U_b,    Ut_b, UU, G4U);
        const int pn = DD * OUTC;
        k_transpose_cvt<<<(pn + 255) / 256, 256, 0, stream>>>(W_proj, Wpt,  DD, OUTC);
    }

    // ---- parallel GEMMs: x projections (+bias) and residual ------------
    {
        // (M/32)*(N/64) waves, 16 waves per block
        const int blocks_proj  = (ROWS / 32) * (G4U / 64) / 16;   // 2048
        const int blocks_resid = (ROWS / 32) * (OUTC / 64) / 16;  // 1024
        k_gemm_bf16<<<blocks_proj,  512, 0, stream>>>(xb, Wt_f, b_f,     zf,    ROWS, G4U);
        k_gemm_bf16<<<blocks_proj,  512, 0, stream>>>(xb, Wt_b, b_b,     zb,    ROWS, G4U);
        k_gemm_bf16<<<blocks_resid, 512, 0, stream>>>(xb, Wpt,  nullptr, resid, ROWS, OUTC);
    }

    // ---- sequential bidirectional scan (214272 B dynamic LDS) ----------
    k_lstm_scan<<<8, 512, SCAN_SMEM, stream>>>(Ut_f, Ut_b, zf, zb, hf, hb);

    // ---- layernorm + residual add --------------------------------------
    k_ln_residual<<<ROWS, 256, 0, stream>>>(hf, hb, resid, gamma, beta, out);
}
